// NetVLAD_23965917511826
// MI455X (gfx1250) — matrix operand order repaired and checked by reading, not compiled, not run on hardware
//
#include <hip/hip_runtime.h>
#include <hip/hip_bf16.h>
#include <stdint.h>

// ---------- problem constants ----------
#define N_IMG 64
#define C_DIM 512
#define P_DIM 1024
#define K_ALL 65
#define K_PAD 80      // 5 tiles of 16 for GEMM1 (softmax needs cluster 64)
#define K_OUT 64      // output clusters (4 tiles of 16 for GEMM2)
#define EPSV  1e-12f

typedef __attribute__((ext_vector_type(16))) _Float16 v16h;
typedef __attribute__((ext_vector_type(8)))  float    v8f;

union FragU { unsigned int u[8]; v16h h; };

// ---------- CDNA5 async global->LDS staging (ASYNCcnt-tracked) ----------
// Generic address of a __shared__ object carries the LDS byte offset in its
// low 32 bits.
__device__ __forceinline__ void async_ld_b128(void* lds_dst, const void* gsrc)
{
    unsigned l = (unsigned)(unsigned long long)lds_dst;
    asm volatile("global_load_async_to_lds_b128 %0, %1, off"
                 :: "v"(l), "v"(gsrc) : "memory");
}
__device__ __forceinline__ void async_wait0()
{
    asm volatile("s_wait_asynccnt 0x0" ::: "memory");
}

// ---------- shared WMMA tile compute: B frag + NT x (A frag, wmma) ----------
template <int NT>
__device__ __forceinline__ void wmma_tiles(const unsigned int* bufA,
                                           const unsigned int* bufB,
                                           v8f* acc, int w, int ml, int hi)
{
    FragU bu;
    #pragma unroll
    for (int e = 0; e < 8; ++e)
        bu.u[e] = bufB[(w * 16 + ml) * 16 + hi * 8 + e];
    v16h bf = bu.h;
    #pragma unroll
    for (int t = 0; t < NT; ++t) {
        FragU au;
        #pragma unroll
        for (int e = 0; e < 8; ++e)
            au.u[e] = bufA[(t * 16 + ml) * 16 + hi * 8 + e];
        acc[t] = __builtin_amdgcn_wmma_f32_16x16x32_f16(
            false, au.h, false, bf, (short)0, acc[t], false, false);
    }
}

// ---------- workspace layout (bytes) ----------
static const size_t OFF_XN    = 0;
static const size_t OFF_A     = OFF_XN   + (size_t)N_IMG*C_DIM*P_DIM*2;
static const size_t OFF_CWH   = OFF_A    + (size_t)N_IMG*K_PAD*P_DIM*2;
static const size_t OFF_ASUM  = OFF_CWH  + (size_t)K_PAD*C_DIM*2;
static const size_t OFF_VLAD  = OFF_ASUM + (size_t)N_IMG*K_PAD*4;
static const size_t OFF_ROWSQ = OFF_VLAD + (size_t)N_IMG*K_OUT*C_DIM*4;

// ============================================================
// K0: conv_w f32 -> f16 (padded to 80 rows), zero rowsq accumulators
// ============================================================
__global__ __launch_bounds__(256)
void prep_kernel(const float* __restrict__ conv_w,
                 _Float16* __restrict__ cwh,
                 float* __restrict__ rowsq)
{
    int idx = blockIdx.x * 256 + threadIdx.x;
    if (idx < K_PAD * C_DIM) {
        int m = idx >> 9;
        int c = idx & (C_DIM - 1);
        float v = (m < K_ALL) ? conv_w[m * C_DIM + c] : 0.0f;
        cwh[idx] = (_Float16)v;
    }
    if (idx < N_IMG * K_OUT) rowsq[idx] = 0.0f;
}

// ============================================================
// K1: per-pixel L2 normalize over channels, f32 -> f16 channel-major
// ============================================================
__global__ __launch_bounds__(256)
void norm_kernel(const float* __restrict__ x, _Float16* __restrict__ xn)
{
    const int n = blockIdx.y;
    const int p = blockIdx.x * 256 + threadIdx.x;
    const float* xb = x + (size_t)n * C_DIM * P_DIM + p;
    float s = 0.0f;
    #pragma unroll 8
    for (int c = 0; c < C_DIM; ++c) {
        float v = xb[(size_t)c * P_DIM];
        s += v * v;
    }
    float inv = 1.0f / fmaxf(sqrtf(s), EPSV);
    _Float16* xo = xn + (size_t)n * C_DIM * P_DIM + p;
    #pragma unroll 8
    for (int c = 0; c < C_DIM; ++c)
        xo[(size_t)c * P_DIM] = (_Float16)(xb[(size_t)c * P_DIM] * inv);
}

// ============================================================
// GEMM1 staging helpers
// ============================================================
__device__ __forceinline__ void g1_stageA_async(unsigned int* bufA,
                                                const _Float16* cwh, int cc, int tid)
{
    for (int q = tid; q < 320; q += 256) {
        int m = q >> 2, o = q & 3;
        async_ld_b128(&bufA[m * 16 + o * 4],
                      cwh + (size_t)m * C_DIM + cc + o * 8);
    }
}
__device__ __forceinline__ void g1_loadB(uint4* breg, const _Float16* xnb,
                                         int cc, int p0, int tid)
{
    #pragma unroll
    for (int i = 0; i < 2; ++i) {
        int q = tid + 256 * i;
        int cr = q >> 4, pc = (q & 15) * 8;
        breg[i] = *(const uint4*)(xnb + (size_t)(cc + cr) * P_DIM + p0 + pc);
    }
}
__device__ __forceinline__ void g1_storeB(const uint4* breg, unsigned int* bufB, int tid)
{
    _Float16* bh = (_Float16*)bufB;
    #pragma unroll
    for (int i = 0; i < 2; ++i) {
        int q = tid + 256 * i;
        int cr = q >> 4, pc = (q & 15) * 8;
        _Float16 hbuf[8];
        *(uint4*)hbuf = breg[i];
        #pragma unroll
        for (int j = 0; j < 8; ++j)
            bh[(pc + j) * 32 + cr] = hbuf[j];
    }
}

// ============================================================
// K2: logits = conv_w(80x512) x xn(512x1024) + bias, softmax over K=65,
// store a (f16, rows>=65 zeroed). Double-buffered LDS: tile i+1 staging
// (async A + global-load B) overlaps WMMA of tile i; B's transpose
// ds_stores run after the WMMAs. One barrier per K-step.
// ============================================================
__global__ __launch_bounds__(256)
void gemm1_softmax_kernel(const _Float16* __restrict__ xn,
                          const _Float16* __restrict__ cwh,
                          const float* __restrict__ conv_b,
                          _Float16* __restrict__ a_out)
{
    __shared__ __align__(16) unsigned int ldsA[2][K_PAD * 16];  // 2 x 80 rows x 32 halves
    __shared__ __align__(16) unsigned int ldsB[2][128 * 16];    // 2 x 128 cols x 32 halves
    __shared__ float ldsBias[K_PAD];

    const int tid = threadIdx.x;
    const int lid = tid & 31;
    const int w   = tid >> 5;
    const int hi  = lid >> 4;
    const int ml  = lid & 15;
    const int n   = blockIdx.y;
    const int p0  = blockIdx.x * 128;

    if (tid < K_PAD) ldsBias[tid] = (tid < K_ALL) ? conv_b[tid] : 0.0f;

    v8f acc[5];
    const v8f zacc = {0.f,0.f,0.f,0.f,0.f,0.f,0.f,0.f};
    #pragma unroll
    for (int t = 0; t < 5; ++t) acc[t] = zacc;

    const _Float16* xnb = xn + (size_t)n * C_DIM * P_DIM;
    const int NSTEP = C_DIM / 32;   // 16

    // prologue: stage step 0 into buffer 0
    g1_stageA_async(ldsA[0], cwh, 0, tid);
    {
        uint4 breg[2];
        g1_loadB(breg, xnb, 0, p0, tid);
        g1_storeB(breg, ldsB[0], tid);
    }

    for (int s = 0; s < NSTEP; ++s) {
        const int cur = s & 1;
        async_wait0();
        __syncthreads();                       // buffer `cur` ready; `cur^1` free
        uint4 breg[2];
        const bool pre = (s + 1 < NSTEP);
        if (pre) {
            g1_loadB(breg, xnb, (s + 1) * 32, p0, tid);   // loads fly during WMMA
            g1_stageA_async(ldsA[cur ^ 1], cwh, (s + 1) * 32, tid);
        }
        wmma_tiles<5>(ldsA[cur], ldsB[cur], acc, w, ml, hi);
        if (pre) g1_storeB(breg, ldsB[cur ^ 1], tid);     // transpose after WMMAs
    }
    __syncthreads();

    // --- fused bias + masked softmax over K (column split across lane^16) ---
    float mx = -3.0e38f;
    #pragma unroll
    for (int t = 0; t < 5; ++t) {
        #pragma unroll
        for (int r = 0; r < 8; ++r) {
            int m = t * 16 + hi * 8 + r;
            if (m < K_ALL) {
                float l = acc[t][r] + ldsBias[m];
                acc[t][r] = l;
                mx = fmaxf(mx, l);
            }
        }
    }
    mx = fmaxf(mx, __shfl_xor(mx, 16, 32));
    float se = 0.0f;
    #pragma unroll
    for (int t = 0; t < 5; ++t) {
        #pragma unroll
        for (int r = 0; r < 8; ++r) {
            int m = t * 16 + hi * 8 + r;
            float e = (m < K_ALL) ? __expf(acc[t][r] - mx) : 0.0f;
            acc[t][r] = e;
            se += e;
        }
    }
    se += __shfl_xor(se, 16, 32);
    float inv = 1.0f / se;

    const int pcol = p0 + w * 16 + ml;
    _Float16* ab = a_out + (size_t)n * K_PAD * P_DIM;
    #pragma unroll
    for (int t = 0; t < 5; ++t) {
        #pragma unroll
        for (int r = 0; r < 8; ++r) {
            int m = t * 16 + hi * 8 + r;
            ab[(size_t)m * P_DIM + pcol] = (_Float16)(acc[t][r] * inv);
        }
    }
}

// ============================================================
// K3: asum[n,k] = sum_p a[n,k,p]. One wave per (n,k) row.
// ============================================================
__global__ __launch_bounds__(256)
void asum_kernel(const _Float16* __restrict__ a_in, float* __restrict__ asum)
{
    const int lid = threadIdx.x & 31;
    const int g = blockIdx.x * 8 + (threadIdx.x >> 5);
    const _Float16* rb = a_in + (size_t)g * P_DIM;
    float s = 0.0f;
    #pragma unroll
    for (int j = 0; j < 4; ++j) {
        uint4 d = *(const uint4*)(rb + j * 256 + lid * 8);
        _Float16 h[8];
        *(uint4*)h = d;
        #pragma unroll
        for (int e = 0; e < 8; ++e) s += (float)h[e];
    }
    s += __shfl_xor(s, 1, 32);
    s += __shfl_xor(s, 2, 32);
    s += __shfl_xor(s, 4, 32);
    s += __shfl_xor(s, 8, 32);
    s += __shfl_xor(s, 16, 32);
    if (lid == 0) asum[g] = s;
}

// ============================================================
// K4: vlad_raw = a(64x1024) x xn^T(1024x512) - asum*centroids.
// Fully async double-buffered staging (3 async b128 per thread per step);
// tile i+1 in flight during WMMA of tile i. One barrier per step.
// ============================================================
__device__ __forceinline__ void g2_stage_async(unsigned int* bufA, unsigned int* bufB,
                                               const _Float16* ab, const _Float16* xnb,
                                               int pk, int c0, int tid)
{
    {   // A tile: 64 rows x 32 halves (256 chunks)
        int q = tid;
        int m = q >> 2, o = q & 3;
        async_ld_b128(&bufA[m * 16 + o * 4],
                      ab + (size_t)m * P_DIM + pk + o * 8);
    }
    #pragma unroll
    for (int i = 0; i < 2; ++i) {   // B tile: 128 rows (c) x 32 halves (512 chunks)
        int q = tid + 256 * i;
        int c = q >> 2, o = q & 3;
        async_ld_b128(&bufB[c * 16 + o * 4],
                      xnb + (size_t)(c0 + c) * P_DIM + pk + o * 8);
    }
}

__global__ __launch_bounds__(256)
void gemm2_vlad_kernel(const _Float16* __restrict__ a_in,
                       const _Float16* __restrict__ xn,
                       const float* __restrict__ asum,
                       const float* __restrict__ centroids,
                       float* __restrict__ vlad_raw,
                       float* __restrict__ rowsq)
{
    __shared__ __align__(16) unsigned int ldsA[2][K_OUT * 16];
    __shared__ __align__(16) unsigned int ldsB[2][128 * 16];

    const int tid = threadIdx.x;
    const int lid = tid & 31;
    const int w   = tid >> 5;
    const int hi  = lid >> 4;
    const int ml  = lid & 15;
    const int n   = blockIdx.y;
    const int c0  = blockIdx.x * 128;

    v8f acc[4];
    const v8f zacc = {0.f,0.f,0.f,0.f,0.f,0.f,0.f,0.f};
    #pragma unroll
    for (int t = 0; t < 4; ++t) acc[t] = zacc;

    const _Float16* ab  = a_in + (size_t)n * K_PAD * P_DIM;
    const _Float16* xnb = xn   + (size_t)n * C_DIM * P_DIM;
    const int NSTEP = P_DIM / 32;   // 32

    g2_stage_async(ldsA[0], ldsB[0], ab, xnb, 0, c0, tid);

    for (int s = 0; s < NSTEP; ++s) {
        const int cur = s & 1;
        async_wait0();
        __syncthreads();
        if (s + 1 < NSTEP)
            g2_stage_async(ldsA[cur ^ 1], ldsB[cur ^ 1], ab, xnb,
                           (s + 1) * 32, c0, tid);
        wmma_tiles<4>(ldsA[cur], ldsB[cur], acc, w, ml, hi);
    }

    // --- fused: subtract asum*centroid, store, accumulate row |.|^2 ---
    const int c = c0 + w * 16 + ml;
    #pragma unroll
    for (int t = 0; t < 4; ++t) {
        #pragma unroll
        for (int r = 0; r < 8; ++r) {
            int m = t * 16 + hi * 8 + r;
            float v = acc[t][r] - asum[n * K_PAD + m] * centroids[m * C_DIM + c];
            vlad_raw[((size_t)n * K_OUT + m) * C_DIM + c] = v;
            float v2 = v * v;
            v2 += __shfl_xor(v2, 1, 32);
            v2 += __shfl_xor(v2, 2, 32);
            v2 += __shfl_xor(v2, 4, 32);
            v2 += __shfl_xor(v2, 8, 32);
            if (ml == 0) atomicAdd(&rowsq[n * K_OUT + m], v2);
        }
    }
}

// ============================================================
// K5: intra-row L2 + global L2 normalize -> output. One block per image.
// ============================================================
__global__ __launch_bounds__(256)
void finalize_kernel(const float* __restrict__ vlad_raw,
                     const float* __restrict__ rowsq,
                     float* __restrict__ out)
{
    __shared__ float invrn[K_OUT];
    __shared__ float fscale;
    const int n = blockIdx.x;
    const int tid = threadIdx.x;

    if (tid < K_OUT)
        invrn[tid] = 1.0f / fmaxf(sqrtf(rowsq[n * K_OUT + tid]), EPSV);
    __syncthreads();
    if (tid == 0) {
        float s = 0.0f;
        for (int k = 0; k < K_OUT; ++k) {
            float ir = invrn[k];
            s += rowsq[n * K_OUT + k] * ir * ir;
        }
        fscale = 1.0f / fmaxf(sqrtf(s), EPSV);
    }
    __syncthreads();
    const size_t base = (size_t)n * K_OUT * C_DIM;
    const float fs = fscale;
    for (int idx = tid; idx < K_OUT * C_DIM; idx += 256) {
        int k = idx >> 9;
        out[base + idx] = vlad_raw[base + idx] * invrn[k] * fs;
    }
}

// ============================================================
extern "C" void kernel_launch(void* const* d_in, const int* in_sizes, int n_in,
                              void* d_out, int out_size, void* d_ws, size_t ws_size,
                              hipStream_t stream)
{
    const float* x         = (const float*)d_in[0];   // (64,512,32,32)
    const float* conv_w    = (const float*)d_in[1];   // (65,512)
    const float* conv_b    = (const float*)d_in[2];   // (65,)
    const float* centroids = (const float*)d_in[3];   // (65,512)

    char* ws = (char*)d_ws;
    _Float16* xn    = (_Float16*)(ws + OFF_XN);
    _Float16* a_ws  = (_Float16*)(ws + OFF_A);
    _Float16* cwh   = (_Float16*)(ws + OFF_CWH);
    float*    asum  = (float*)   (ws + OFF_ASUM);
    float*    vlad  = (float*)   (ws + OFF_VLAD);
    float*    rowsq = (float*)   (ws + OFF_ROWSQ);
    float*    out   = (float*)d_out;

    prep_kernel<<<(K_PAD * C_DIM + 255) / 256, 256, 0, stream>>>(conv_w, cwh, rowsq);
    norm_kernel<<<dim3(P_DIM / 256, N_IMG), 256, 0, stream>>>(x, xn);
    gemm1_softmax_kernel<<<dim3(P_DIM / 128, N_IMG), 256, 0, stream>>>(xn, cwh, conv_b, a_ws);
    asum_kernel<<<(N_IMG * K_PAD) / 8, 256, 0, stream>>>(a_ws, asum);
    gemm2_vlad_kernel<<<dim3(C_DIM / 128, N_IMG), 256, 0, stream>>>(
        a_ws, xn, asum, centroids, vlad, rowsq);
    finalize_kernel<<<N_IMG, 256, 0, stream>>>(vlad, rowsq, out);
}
